// CustomCorrelationSampler_32272384262701
// MI455X (gfx1250) — matrix operand order repaired and checked by reading, not compiled, not run on hardware
//
#include <hip/hip_runtime.h>
#include <hip/hip_bf16.h>

// Problem constants (match reference setup_inputs)
#define BB 16
#define CC 256
#define HH 96
#define WW 160
#define PATCH 9
#define NPLANE 81          // 9x9 displacements
#define YB 24              // output-row block per workgroup
#define NYB (HH / YB)      // 4
#define XT (WW / 16)       // 10 x-tiles
#define AK 40              // padded K-stride in bf16 elems (80B -> bank-conflict free)
#define NEG_SLOPE 0.1f

typedef __attribute__((ext_vector_type(16))) __bf16       v16bf;
typedef __attribute__((ext_vector_type(8)))  float        v8f;
typedef __attribute__((ext_vector_type(4)))  unsigned int v4u;

static __device__ __forceinline__ int imax(int a, int b) { return a > b ? a : b; }
static __device__ __forceinline__ int imin(int a, int b) { return a < b ? a : b; }

// Assemble a v16bf WMMA fragment from two 16-byte LDS chunks (2x ds_load_b128).
static __device__ __forceinline__ v16bf frag2(const __bf16* p0, const __bf16* p1) {
  union { v4u q[2]; v16bf v; } u;
  u.q[0] = *(const v4u*)p0;
  u.q[1] = *(const v4u*)p1;
  return u.v;
}

// fp32 -> bf16 hi/lo split: x ~= h + l, recombined to ~2^-17 relative via 3 WMMAs.
static __device__ __forceinline__ void split2(float x, __bf16& h, __bf16& l) {
  h = (__bf16)x;
  l = (__bf16)(x - (float)h);
}

__global__ __launch_bounds__(256, 1)
void corr_volume_wmma_kernel(const float* __restrict__ f1,
                             const float* __restrict__ f2,
                             float* __restrict__ out) {
  // A ring: 5 f1 row-tiles [slot][kc][hi/lo][M=16][AK]   -> 100 KB
  // B strip: current f2 row [kc][hi/lo][col=24][AK]      ->  30 KB
  __shared__ __align__(16) __bf16 sA[5][8][2][16][AK];
  __shared__ __align__(16) __bf16 sB[8][2][24][AK];

  const int t     = threadIdx.x;
  const int lane  = t & 31;
  const int wave  = t >> 5;
  const int ntile = wave >> 2;   // 0: x2 cols [x0-4, x0+11], 1: [x0+4, x0+19]
  const int jmod  = wave & 3;    // y-in-window assignment
  const int n16   = lane & 15;
  const int khalf = lane >> 4;

  int bid = blockIdx.x;
  const int type = bid & 1; bid >>= 1;       // dy-half: 0 -> dy in [-4,0], 1 -> dy in [1,4]
  const int xt = bid % XT;  bid /= XT;
  const int yb = bid % NYB; bid /= NYB;
  const int b  = bid;

  const int x0  = xt * 16;
  const int y0  = yb * YB;
  const int y1  = y0 + YB;
  const int wlo = type ? -4 : 0;             // window: y in [r+wlo, r+whi]
  const int whi = type ? -1 : 4;

  const int rBeg = imax(0, y0 - whi);
  const int rEnd = imin(HH, y1 - wlo);

  const size_t fbase = (size_t)b * CC * HH * WW;

  // ---- loop-invariant per-thread load mappings ----
  const int  aXi  = t & 15;                  // f1: x within tile
  const int  aC0  = t >> 4;                  // f1: channel base, step 16
  const int  bCol = t & 31;                  // f2: strip column (cols >= 24 idle)
  const int  bC0  = t >> 5;                  // f2: channel base, step 8
  const bool bAct = bCol < 24;
  const int  bX2  = x0 - 4 + bCol;
  const bool bInW = bAct && (bX2 >= 0) && (bX2 < WW);

  float rgA[16];                             // staged f1 row (register double buffer)
  float rgB[32];                             // staged f2 row strip

  auto gloadA = [&](int row) {
#pragma unroll
    for (int i = 0; i < 16; ++i) {
      const int c = aC0 + 16 * i;
      rgA[i] = f1[fbase + ((size_t)c * HH + row) * WW + (x0 + aXi)];
    }
  };
  auto storeA = [&](int slot) {              // commit staged regs -> bf16 hi/lo in LDS
    __bf16* dst = &sA[slot][0][0][0][0];
#pragma unroll
    for (int i = 0; i < 16; ++i) {
      const int c = aC0 + 16 * i;
      __bf16 h, l; split2(rgA[i], h, l);
      const int kc = c >> 5, kk = c & 31;
      dst[(size_t)((kc * 2 + 0) * 16 + aXi) * AK + kk] = h;
      dst[(size_t)((kc * 2 + 1) * 16 + aXi) * AK + kk] = l;
    }
  };
  auto gloadB = [&](int r) {
#pragma unroll
    for (int i = 0; i < 32; ++i) {
      const int c = bC0 + 8 * i;
      rgB[i] = bInW ? f2[fbase + ((size_t)c * HH + r) * WW + bX2] : 0.0f;
    }
  };
  auto storeB = [&]() {
    if (!bAct) return;
    __bf16* dst = &sB[0][0][0][0];
#pragma unroll
    for (int i = 0; i < 32; ++i) {
      const int c = bC0 + 8 * i;
      __bf16 h, l; split2(rgB[i], h, l);
      const int kc = c >> 5, kk = c & 31;
      dst[(size_t)((kc * 2 + 0) * 24 + bCol) * AK + kk] = h;
      dst[(size_t)((kc * 2 + 1) * 24 + bCol) * AK + kk] = l;
    }
  };

  // Prologue: directly load f1 rows [rBeg+wlo, rBeg+whi-1] that fall in the y-block.
  for (int row = rBeg + wlo; row < rBeg + whi; ++row)
    if (row >= y0 && row < y1) { gloadA(row); storeA(row % 5); }

  // Stage the first iteration's rows.
  bool haveA;
  {
    const int nr = rBeg + whi;
    haveA = (nr >= y0 && nr < y1);
    if (haveA) gloadA(nr);
    gloadB(rBeg);
  }

  int srow = ((rBeg + wlo) % 5 + 5) % 5;     // ring slot of row (r + wlo)
  int snew = (rBeg + whi) % 5;               // ring slot of row (r + whi), nonneg by construction
  const int jmax = whi - wlo;                // 4 (type0) or 3 (type1)

  for (int r = rBeg; r < rEnd; ++r) {
    // Commit staged registers (loaded during previous iteration) into LDS.
    if (haveA) storeA(snew);
    storeB();

    // Issue next iteration's global loads now: they overlap the WMMA phase below;
    // their s_wait lands at the next iteration's commit.
    bool haveAn = false;
    if (r + 1 < rEnd) {
      const int nr = r + 1 + whi;
      haveAn = (nr >= y0 && nr < y1);
      if (haveAn) gloadA(nr);
      gloadB(r + 1);
      if (r + 2 < rEnd && bAct)              // warm r+2 in L2 (global_prefetch_b8)
        __builtin_prefetch(&f2[fbase + ((size_t)bC0 * HH + (r + 2)) * WW + bX2], 0, 1);
    }
    __syncthreads();

    for (int j = jmod; j <= jmax; j += 4) {  // wave-uniform control: EXEC stays full for WMMA
      const int y = r + wlo + j;
      if (y < y0 || y >= y1) continue;
      int slot = srow + j; if (slot >= 5) slot -= 5;   // no integer division in hot loop

      // Per-lane fragment base pointers.
      // A (16x32 bf16, M x K): lane M=n16; khalf selects K {0..7,16..23} vs {8..15,24..31}.
      const __bf16* Ab = &sA[slot][0][0][0][0] + (size_t)n16 * AK + (khalf ? 8 : 0);
      // B (32x16 bf16, K x N): lane N=n16 (+8 cols for ntile1); khalf selects K {0..7,16..23} vs {8..15,24..31}.
      const __bf16* Bb = &sB[0][0][0][0] + (size_t)(ntile * 8 + n16) * AK + khalf * 8;

      v8f acc = {};
#pragma unroll
      for (int kc = 0; kc < 8; ++kc) {
        const __bf16* a0 = Ab + (size_t)(kc * 2 + 0) * 16 * AK;
        const __bf16* a1 = Ab + (size_t)(kc * 2 + 1) * 16 * AK;
        const __bf16* b0 = Bb + (size_t)(kc * 2 + 0) * 24 * AK;
        const __bf16* b1 = Bb + (size_t)(kc * 2 + 1) * 24 * AK;
        const v16bf ah = frag2(a0, a0 + 16);
        const v16bf al = frag2(a1, a1 + 16);
        const v16bf bh = frag2(b0, b0 + 16);
        const v16bf bl = frag2(b1, b1 + 16);
        // Split-precision accumulation: a*b ~= ah*bh + ah*bl + al*bh (fp32 accum).
        acc = __builtin_amdgcn_wmma_f32_16x16x32_bf16(false, ah, false, bh, (short)0, acc, false, false);
        acc = __builtin_amdgcn_wmma_f32_16x16x32_bf16(false, ah, false, bl, (short)0, acc, false, false);
        acc = __builtin_amdgcn_wmma_f32_16x16x32_bf16(false, al, false, bh, (short)0, acc, false, false);
      }

      // Scatter the 9-wide diagonal band to out[b][dyi*9+dxi][y][x], leaky-ReLU applied.
      const int dyi = (r - y) + 4;           // reference plane index i
#pragma unroll
      for (int v = 0; v < 8; ++v) {
        const int m   = v + (khalf ? 8 : 0); // C/D layout: lanes16-31 hold M = v+8
        const int dxi = n16 - m + (ntile ? 8 : 0);
        const bool wr = (dxi >= 0) && (dxi <= 8) && (ntile == 0 || n16 >= 8); // dedupe overlap
        if (wr) {
          float val = acc[v];
          val = val > 0.0f ? val : val * NEG_SLOPE;
          const int plane = dyi * PATCH + dxi;
          out[(((size_t)b * NPLANE + plane) * HH + y) * WW + (x0 + m)] = val;
        }
      }
    }
    __syncthreads();

    haveA = haveAn;
    srow = (srow + 1 == 5) ? 0 : srow + 1;
    snew = (snew + 1 == 5) ? 0 : snew + 1;
  }
}

extern "C" void kernel_launch(void* const* d_in, const int* in_sizes, int n_in,
                              void* d_out, int out_size, void* d_ws, size_t ws_size,
                              hipStream_t stream) {
  (void)in_sizes; (void)n_in; (void)d_ws; (void)ws_size;
  const float* f1 = (const float*)d_in[0];
  const float* f2 = (const float*)d_in[1];
  float* out = (float*)d_out;

  // Out-of-range (y+dy) planes are exactly zero in the reference; pre-zero the output.
  hipMemsetAsync(d_out, 0, (size_t)out_size * sizeof(float), stream);

  const int nblocks = BB * NYB * XT * 2;     // 16 * 4 * 10 * 2 = 1280 workgroups
  corr_volume_wmma_kernel<<<dim3(nblocks), dim3(256), 0, stream>>>(f1, f2, out);
}
